// ConservativeFluxOperator1D_39994735460750
// MI455X (gfx1250) — compile-verified, matrix operand order checked by reading, not run on hardware
//
#include <hip/hip_runtime.h>
#include <hip/hip_bf16.h>
#include <math.h>

typedef __attribute__((ext_vector_type(16))) _Float16 v16h;
typedef __attribute__((ext_vector_type(8)))  _Float16 v8h;
typedef __attribute__((ext_vector_type(8)))  float    v8f;

#define WPB      4      // waves per block (wave32)
#define LATENT   32
#define HIDDEN   64
#define TAU_DIM  16

// ---------------------------------------------------------------------------
// WMMA helpers (CDNA5 gfx1250: D = A(16x32 f16) * B(32x16 f16) + C(16x16 f32))
// ---------------------------------------------------------------------------
__device__ __forceinline__ v8f wmma16(v16h a, v16h b, v8f c) {
  return __builtin_amdgcn_wmma_f32_16x16x32_f16(false, a, false, b, (short)0, c,
                                                false, false);
}

// Branch-free GELU (tanh form): x * sigmoid(1.5957691*(x + 0.044715 x^3)).
// One v_exp_f32 + one v_rcp_f32; no exec-mask churn around the WMMAs.
__device__ __forceinline__ float gelu_fast(float x) {
  float x3 = x * x * x;
  // t = log2(e) * 1.5957691 * (x + 0.044715 x^3)
  float t = 2.302118f * x + 0.1029364f * x3;
  float e = __builtin_amdgcn_exp2f(-t);           // exp(-1.59577(x+0.044715x^3))
  return x * __builtin_amdgcn_rcpf(1.0f + e);     // x * sigmoid(...)
}

// Branch-free tanh: 1 - 2/(exp2(2*log2e*x) + 1)
__device__ __forceinline__ float tanh_fast(float x) {
  float e = __builtin_amdgcn_exp2f(2.885390082f * x);  // e^(2x)
  return 1.0f - 2.0f * __builtin_amdgcn_rcpf(e + 1.0f);
}

// B fragments are pre-shuffled by prep_weights into contiguous per-lane v16h.
__device__ __forceinline__ v16h load_bfrag(const _Float16* w, int kt, int nt,
                                           int NT, int lane) {
  return *(const v16h*)(w + ((((kt * NT) + nt) * 32 + lane) << 4));
}

// Gather an A fragment from a row-major f16 LDS buffer.
// A layout (16-bit 16x32): lanes 0-15 -> M=lane, K {0..7,16..23};
//                          lanes 16-31 -> M=lane-16, K {8..15,24..31}.
__device__ __forceinline__ v16h lds_afrag(const _Float16* buf, int stride,
                                          int rowOff, int kt, int lane) {
  int row = rowOff + (lane & 15);
  int klo = (lane & 16) ? 8 : 0;
  const _Float16* p = buf + row * stride + kt * 32 + klo;
  v8h lo = *(const v8h*)(p);
  v8h hi = *(const v8h*)(p + 16);
  v16h a;
#pragma unroll
  for (int i = 0; i < 8; ++i) { a[i] = lo[i]; a[8 + i] = hi[i]; }
  return a;
}

// Scatter a D tile (f32 accumulator) into a row-major f16 LDS buffer,
// applying bias (indexed by output column) and optional fast GELU.
// D layout: VGPR r, lanes 0-15 -> (M=r, N=lane); lanes 16-31 -> (M=8+r, N=lane-16).
__device__ __forceinline__ void store_tile(_Float16* buf, int stride, int rowOff,
                                           int nt, v8f d, const float* bias,
                                           bool act, int lane) {
  int col = nt * 16 + (lane & 15);
  float bv = bias[col];
  int rbase = rowOff + ((lane & 16) ? 8 : 0);
#pragma unroll
  for (int r = 0; r < 8; ++r) {
    float v = d[r] + bv;
    buf[(rbase + r) * stride + col] = (_Float16)(act ? gelu_fast(v) : v);
  }
}

// ---------------------------------------------------------------------------
// Weight repack: f32 [Kact x Nact] row-major -> f16 fragment-ordered,
// zero-padded to [Kpad x Npad].  Element h of lane's v16h holds
// K = kt*32 + klo + (h<8 ? h : h+8), matching the A-side gather above.
// ---------------------------------------------------------------------------
__global__ void prep_weights(const float* __restrict__ W, _Float16* __restrict__ dst,
                             int Kact, int Nact, int Kpad, int Npad) {
  int i = blockIdx.x * blockDim.x + threadIdx.x;
  int total = Kpad * Npad;
  if (i >= total) return;
  int h = i & 15;
  int lane = (i >> 4) & 31;
  int frag = i >> 9;
  int NT = Npad >> 4;
  int nt = frag % NT;
  int kt = frag / NT;
  int klo = (lane & 16) ? 8 : 0;
  int K = kt * 32 + klo + (h < 8 ? h : h + 8);
  int n = nt * 16 + (lane & 15);
  float v = (K < Kact && n < Nact) ? W[K * Nact + n] : 0.0f;
  dst[i] = (_Float16)v;
}

// ---------------------------------------------------------------------------
// Per-batch tau MLP: 1 -> 64 -> 64 -> 16 (tiny; plain VALU, one block/batch)
// ---------------------------------------------------------------------------
__global__ void tau_mlp(const float* __restrict__ tau,
                        const float* W1, const float* b1,
                        const float* W2, const float* b2,
                        const float* W3, const float* b3,
                        _Float16* __restrict__ tauEmb) {
  __shared__ float h1[HIDDEN], h2[HIDDEN];
  int b = blockIdx.x, j = threadIdx.x;
  float t = tau[b];
  h1[j] = gelu_fast(W1[j] * t + b1[j]);
  __syncthreads();
  float a = b2[j];
  for (int k = 0; k < HIDDEN; ++k) a += W2[k * HIDDEN + j] * h1[k];
  h2[j] = gelu_fast(a);
  __syncthreads();
  if (j < TAU_DIM) {
    float o = b3[j];
    for (int k = 0; k < HIDDEN; ++k) o += W3[k * TAU_DIM + j] * h2[k];
    tauEmb[b * TAU_DIM + j] = (_Float16)o;
  }
}

// ---------------------------------------------------------------------------
// Pass 1: msg MLP (left+right, M=32 stacked) + node MLP, residual -> z_new
// One wave handles 16 consecutive nodes of one batch row.
// ---------------------------------------------------------------------------
__global__ void __launch_bounds__(WPB * 32)
pass1_msg_node(const float* __restrict__ u_t, const float* __restrict__ encW,
               const _Float16* wM1, const _Float16* wM2, const _Float16* wM3,
               const float* bM1, const float* bM2, const float* bM3,
               const _Float16* wN1, const _Float16* wN2, const _Float16* wN3,
               const float* bN1, const float* bN2, const float* bN3,
               _Float16* __restrict__ zNew, int B, int N) {
  __shared__ __align__(16) _Float16 sBuf[WPB][2][32 * 64];
  __shared__ float sU[WPB][18];
  __shared__ float sE[32];

  const int lane = threadIdx.x & 31;
  const int w = threadIdx.x >> 5;
  const int wave = blockIdx.x * WPB + w;
  const int tilesPerB = N >> 4;
  const int totalWaves = B * tilesPerB;
  const bool valid = wave < totalWaves;
  const int wv = valid ? wave : 0;
  const int b = wv / tilesPerB;
  const int s = (wv % tilesPerB) << 4;

  if (threadIdx.x < 32)
    sE[threadIdx.x] = (threadIdx.x == 0) ? 1.0f : encW[threadIdx.x - 1];
  if (lane < 18) {
    int idx = s + lane - 1;
    idx = idx < 0 ? 0 : (idx > N - 1 ? N - 1 : idx);
    sU[w][lane] = u_t[(size_t)b * N + idx];
  }
  __syncthreads();

  const int row = lane & 15;
  const int klo = (lane & 16) ? 8 : 0;
  const float uc = sU[w][row + 1];
  const float ul = sU[w][row + 0];
  const float ur = sU[w][row + 2];

  float eK[16];
#pragma unroll
  for (int h = 0; h < 16; ++h) eK[h] = sE[klo + (h < 8 ? h : h + 8)];

  _Float16* bufA = sBuf[w][0];
  _Float16* bufB = sBuf[w][1];

  // msg L1 A-fragments built in registers: z = u*e is rank-1 in the scalar u.
  v16h aC, aL, aR, aD0, aD1;
#pragma unroll
  for (int h = 0; h < 16; ++h) {
    aC[h] = (_Float16)(uc * eK[h]);
    aL[h] = (_Float16)(ul * eK[h]);
    aR[h] = (_Float16)(ur * eK[h]);
    aD0[h] = (_Float16)0.f;
    aD1[h] = (_Float16)0.f;
  }
  if (klo == 0) { aD0[0] = (_Float16)(-1.f); aD1[0] = (_Float16)(1.f); }

  // ---- msg L1: M=32 (rows 0-15 left, 16-31 right), K=96 (65 used), N=64 ----
#pragma unroll
  for (int mt = 0; mt < 2; ++mt) {
    v16h aN = mt ? aR : aL;
    v16h aD = mt ? aD1 : aD0;
#pragma unroll
    for (int nt = 0; nt < 4; ++nt) {
      v8f c = {};
      c = wmma16(aC, load_bfrag(wM1, 0, nt, 4, lane), c);
      c = wmma16(aN, load_bfrag(wM1, 1, nt, 4, lane), c);
      c = wmma16(aD, load_bfrag(wM1, 2, nt, 4, lane), c);
      store_tile(bufA, 64, mt * 16, nt, c, bM1, true, lane);
    }
  }
  __syncthreads();

  // ---- msg L2: 64 -> 64, GELU ----
#pragma unroll
  for (int mt = 0; mt < 2; ++mt)
#pragma unroll
    for (int nt = 0; nt < 4; ++nt) {
      v8f c = {};
#pragma unroll
      for (int kt = 0; kt < 2; ++kt)
        c = wmma16(lds_afrag(bufA, 64, mt * 16, kt, lane),
                   load_bfrag(wM2, kt, nt, 4, lane), c);
      store_tile(bufB, 64, mt * 16, nt, c, bM2, true, lane);
    }
  __syncthreads();

  // ---- msg L3: 64 -> 32, no activation; m_left+m_right is a register add ----
  v8f msum[2];
#pragma unroll
  for (int nt = 0; nt < 2; ++nt) {
    v8f c0 = {}, c1 = {};
#pragma unroll
    for (int kt = 0; kt < 2; ++kt) {
      v16h bf = load_bfrag(wM3, kt, nt, 2, lane);
      c0 = wmma16(lds_afrag(bufB, 64, 0, kt, lane), bf, c0);
      c1 = wmma16(lds_afrag(bufB, 64, 16, kt, lane), bf, c1);
    }
    float bv = 2.0f * bM3[nt * 16 + (lane & 15)];
#pragma unroll
    for (int r = 0; r < 8; ++r) msum[nt][r] = c0[r] + c1[r] + bv;
  }
  __syncthreads();

  // ---- build node-MLP input [z | m_sum] (16 x 64) in LDS ----
  {
    int ks = (lane & 16) ? 16 : 0;
#pragma unroll
    for (int k = 0; k < 16; ++k)
      bufA[row * 64 + ks + k] = (_Float16)(uc * sE[ks + k]);
    int rb = (lane & 16) ? 8 : 0;
#pragma unroll
    for (int nt = 0; nt < 2; ++nt) {
      int col = 32 + nt * 16 + (lane & 15);
#pragma unroll
      for (int r = 0; r < 8; ++r)
        bufA[(rb + r) * 64 + col] = (_Float16)msum[nt][r];
    }
  }
  __syncthreads();

  // ---- node L1 ----
#pragma unroll
  for (int nt = 0; nt < 4; ++nt) {
    v8f c = {};
#pragma unroll
    for (int kt = 0; kt < 2; ++kt)
      c = wmma16(lds_afrag(bufA, 64, 0, kt, lane),
                 load_bfrag(wN1, kt, nt, 4, lane), c);
    store_tile(bufB, 64, 0, nt, c, bN1, true, lane);
  }
  __syncthreads();

  // ---- node L2 ----
#pragma unroll
  for (int nt = 0; nt < 4; ++nt) {
    v8f c = {};
#pragma unroll
    for (int kt = 0; kt < 2; ++kt)
      c = wmma16(lds_afrag(bufB, 64, 0, kt, lane),
                 load_bfrag(wN2, kt, nt, 4, lane), c);
    store_tile(bufA, 64, 0, nt, c, bN2, true, lane);
  }
  __syncthreads();

  // ---- node L3: 64 -> 32, residual z + out, store f16 z_new ----
  int rb = (lane & 16) ? 8 : 0;
#pragma unroll
  for (int nt = 0; nt < 2; ++nt) {
    v8f c = {};
#pragma unroll
    for (int kt = 0; kt < 2; ++kt)
      c = wmma16(lds_afrag(bufA, 64, 0, kt, lane),
                 load_bfrag(wN3, kt, nt, 2, lane), c);
    int nl = nt * 16 + (lane & 15);
    float bv = bN3[nl];
    float ev = sE[nl];
    if (valid) {
#pragma unroll
      for (int r = 0; r < 8; ++r) {
        int m = rb + r;
        float zv = sU[w][m + 1] * ev + c[r] + bv;
        zNew[((size_t)(b * N + s + m)) * LATENT + nl] = (_Float16)zv;
      }
    }
  }
}

// ---------------------------------------------------------------------------
// Pass 2: flux MLP per edge (K=96: z_prev|z_cur|tau_emb), tanh -> G
// One wave handles 16 consecutive edges of one batch row (N+1 edges total).
// ---------------------------------------------------------------------------
__global__ void __launch_bounds__(WPB * 32)
pass2_flux(const _Float16* __restrict__ zNew, const _Float16* __restrict__ tauEmb,
           const _Float16* wF1, const _Float16* wF2, const _Float16* wF3,
           const float* bF1, const float* bF2, const float* bF3,
           float* __restrict__ G, int B, int N) {
  __shared__ __align__(16) _Float16 sBuf[WPB][2][16 * 64];
  __shared__ float sG[WPB][16];

  const int lane = threadIdx.x & 31;
  const int w = threadIdx.x >> 5;
  const int ET = (N + 16) >> 4;  // edge tiles per row, covers N+1 edges
  const int wave = blockIdx.x * WPB + w;
  const int totalWaves = B * ET;
  const bool valid = wave < totalWaves;
  const int wv = valid ? wave : 0;
  const int b = wv / ET;
  const int e0 = (wv % ET) << 4;

  const int row = lane & 15;
  const int klo = (lane & 16) ? 8 : 0;
  const int e = e0 + row;
  int ip = e - 1;
  ip = ip < 0 ? 0 : (ip > N - 1 ? N - 1 : ip);
  int ic = e > N - 1 ? N - 1 : e;

  const _Float16* zp = zNew + (size_t)(b * N + ip) * LATENT + klo;
  const _Float16* zc = zNew + (size_t)(b * N + ic) * LATENT + klo;
  v16h aP, aC, aT;
  {
    v8h lo = *(const v8h*)zp, hi = *(const v8h*)(zp + 16);
#pragma unroll
    for (int i = 0; i < 8; ++i) { aP[i] = lo[i]; aP[8 + i] = hi[i]; }
  }
  {
    v8h lo = *(const v8h*)zc, hi = *(const v8h*)(zc + 16);
#pragma unroll
    for (int i = 0; i < 8; ++i) { aC[i] = lo[i]; aC[8 + i] = hi[i]; }
  }
  {
    const _Float16* tp = tauEmb + b * TAU_DIM + klo;
#pragma unroll
    for (int i = 0; i < 8; ++i) { aT[i] = tp[i]; aT[8 + i] = (_Float16)0.f; }
  }

  _Float16* bufA = sBuf[w][0];
  _Float16* bufB = sBuf[w][1];

  // ---- flux L1: K=96 (80 used) -> 64, GELU ----
#pragma unroll
  for (int nt = 0; nt < 4; ++nt) {
    v8f c = {};
    c = wmma16(aP, load_bfrag(wF1, 0, nt, 4, lane), c);
    c = wmma16(aC, load_bfrag(wF1, 1, nt, 4, lane), c);
    c = wmma16(aT, load_bfrag(wF1, 2, nt, 4, lane), c);
    store_tile(bufA, 64, 0, nt, c, bF1, true, lane);
  }
  __syncthreads();

  // ---- flux L2: 64 -> 64, GELU ----
#pragma unroll
  for (int nt = 0; nt < 4; ++nt) {
    v8f c = {};
#pragma unroll
    for (int kt = 0; kt < 2; ++kt)
      c = wmma16(lds_afrag(bufA, 64, 0, kt, lane),
                 load_bfrag(wF2, kt, nt, 4, lane), c);
    store_tile(bufB, 64, 0, nt, c, bF2, true, lane);
  }
  __syncthreads();

  // ---- flux L3: 64 -> 1 (N padded to 16; column 0 real) ----
  {
    v8f c = {};
#pragma unroll
    for (int kt = 0; kt < 2; ++kt)
      c = wmma16(lds_afrag(bufB, 64, 0, kt, lane),
                 load_bfrag(wF3, kt, 0, 1, lane), c);
    if ((lane & 15) == 0) {
      int rb = (lane & 16) ? 8 : 0;
#pragma unroll
      for (int r = 0; r < 8; ++r) sG[w][rb + r] = c[r];
    }
  }
  __syncthreads();
  if (lane < 16) {
    int eo = e0 + lane;
    if (valid && eo <= N)
      G[(size_t)b * (N + 1) + eo] = tanh_fast(sG[w][lane] + bF3[0]);  // FLUX_SCALE=1
  }
}

// ---------------------------------------------------------------------------
// Pass 3: u_next = u - (G[n+1] - G[n]) / dx
// ---------------------------------------------------------------------------
__global__ void pass3_update(const float* __restrict__ u_t, const float* __restrict__ G,
                             const float* __restrict__ dxp, float* __restrict__ out,
                             int B, int N) {
  int i = blockIdx.x * blockDim.x + threadIdx.x;
  if (i >= B * N) return;
  int b = i / N, n = i - b * N;
  float dx = dxp[0];
  const float* g = G + (size_t)b * (N + 1);
  out[i] = u_t[i] - (g[n + 1] - g[n]) / dx;
}

// ---------------------------------------------------------------------------
// Host launcher
// ---------------------------------------------------------------------------
extern "C" void kernel_launch(void* const* d_in, const int* in_sizes, int n_in,
                              void* d_out, int out_size, void* d_ws, size_t ws_size,
                              hipStream_t stream) {
  const float* u_t  = (const float*)d_in[0];
  const float* tau  = (const float*)d_in[1];
  const float* dxp  = (const float*)d_in[2];
  const float* encW = (const float*)d_in[3];
  const float* tW1 = (const float*)d_in[4],  *tb1 = (const float*)d_in[5];
  const float* tW2 = (const float*)d_in[6],  *tb2 = (const float*)d_in[7];
  const float* tW3 = (const float*)d_in[8],  *tb3 = (const float*)d_in[9];
  const float* mW1 = (const float*)d_in[10], *mb1 = (const float*)d_in[11];
  const float* mW2 = (const float*)d_in[12], *mb2 = (const float*)d_in[13];
  const float* mW3 = (const float*)d_in[14], *mb3 = (const float*)d_in[15];
  const float* nW1 = (const float*)d_in[16], *nb1 = (const float*)d_in[17];
  const float* nW2 = (const float*)d_in[18], *nb2 = (const float*)d_in[19];
  const float* nW3 = (const float*)d_in[20], *nb3 = (const float*)d_in[21];
  const float* fW1 = (const float*)d_in[22], *fb1 = (const float*)d_in[23];
  const float* fW2 = (const float*)d_in[24], *fb2 = (const float*)d_in[25];
  const float* fW3 = (const float*)d_in[26], *fb3 = (const float*)d_in[27];
  (void)n_in; (void)ws_size;

  const int B = in_sizes[1];
  const int N = in_sizes[0] / B;

  char* ws = (char*)d_ws;
  size_t off = 0;
  auto take = [&](size_t bytes) {
    size_t o = off;
    off = (off + bytes + 255) & ~(size_t)255;
    return o;
  };
  size_t oM1 = take(96 * 64 * 2), oM2 = take(64 * 64 * 2), oM3 = take(64 * 32 * 2);
  size_t oN1 = take(64 * 64 * 2), oN2 = take(64 * 64 * 2), oN3 = take(64 * 32 * 2);
  size_t oF1 = take(96 * 64 * 2), oF2 = take(64 * 64 * 2), oF3 = take(64 * 16 * 2);
  size_t oTe = take((size_t)B * TAU_DIM * 2);
  size_t oZ  = take((size_t)B * N * LATENT * 2);
  size_t oG  = take((size_t)B * (N + 1) * 4);

  auto H = [&](size_t o) { return (_Float16*)(ws + o); };

  // Weight repack (fragment-ordered f16, K/N zero-padded).
  auto prep = [&](const float* W, size_t o, int Ka, int Na, int Kp, int Np) {
    int total = Kp * Np;
    prep_weights<<<(total + 255) / 256, 256, 0, stream>>>(W, H(o), Ka, Na, Kp, Np);
  };
  prep(mW1, oM1, 2 * LATENT + 1, HIDDEN, 96, 64);
  prep(mW2, oM2, HIDDEN, HIDDEN, 64, 64);
  prep(mW3, oM3, HIDDEN, LATENT, 64, 32);
  prep(nW1, oN1, 2 * LATENT, HIDDEN, 64, 64);
  prep(nW2, oN2, HIDDEN, HIDDEN, 64, 64);
  prep(nW3, oN3, HIDDEN, LATENT, 64, 32);
  prep(fW1, oF1, 2 * LATENT + TAU_DIM, HIDDEN, 96, 64);
  prep(fW2, oF2, HIDDEN, HIDDEN, 64, 64);
  prep(fW3, oF3, HIDDEN, 1, 64, 16);

  tau_mlp<<<B, HIDDEN, 0, stream>>>(tau, tW1, tb1, tW2, tb2, tW3, tb3, H(oTe));

  int waves1 = B * (N / 16);
  pass1_msg_node<<<(waves1 + WPB - 1) / WPB, WPB * 32, 0, stream>>>(
      u_t, encW, H(oM1), H(oM2), H(oM3), mb1, mb2, mb3,
      H(oN1), H(oN2), H(oN3), nb1, nb2, nb3, H(oZ), B, N);

  int waves2 = B * ((N + 16) / 16);
  pass2_flux<<<(waves2 + WPB - 1) / WPB, WPB * 32, 0, stream>>>(
      H(oZ), H(oTe), H(oF1), H(oF2), H(oF3), fb1, fb2, fb3,
      (float*)(ws + oG), B, N);

  pass3_update<<<(B * N + 255) / 256, 256, 0, stream>>>(
      u_t, (const float*)(ws + oG), dxp, (float*)d_out, B, N);
}
